// TransformerBlock_63617055589128
// MI455X (gfx1250) — compile-verified, matrix operand order checked by reading, not run on hardware
//
#include <hip/hip_runtime.h>

typedef __attribute__((ext_vector_type(16))) __bf16 v16bf;
typedef __attribute__((ext_vector_type(4)))  __bf16 v4bf;
typedef __attribute__((ext_vector_type(8)))  float  v8f;

#if __has_builtin(__builtin_nontemporal_load)
#define NT_LOAD(p) __builtin_nontemporal_load(p)
#else
#define NT_LOAD(p) (*(p))
#endif

// ---------------------------------------------------------------------------
// WMMA helper: D = A(16x32 bf16) x B(32x16 bf16) + C(16x16 f32)
// ---------------------------------------------------------------------------
__device__ inline v8f wmma_bf16(v16bf a, v16bf b, v8f c) {
  return __builtin_amdgcn_wmma_f32_16x16x32_bf16(
      /*neg_a=*/false, a, /*neg_b=*/false, b,
      /*c_mod=*/(short)0, c, /*reuse_a=*/false, /*reuse_b=*/false);
}

// Load one 16x32 bf16 fragment (A-layout, or B-layout where lane = column and
// the source is stored column-contiguous, i.e. [N][K] row-major).
__device__ inline v16bf frag_ld(const __bf16* base, int stride) {
  int lane = threadIdx.x & 31;
  int r    = lane & 15;
  int h8   = (lane >> 4) * 8;
  const __bf16* p = base + (size_t)r * stride + h8;
  union { uint4 u[2]; v16bf v; } u;
  u.u[0] = *(const uint4*)(p);
  u.u[1] = *(const uint4*)(p + 16);
  return u.v;
}

// ---------------------------------------------------------------------------
// 16-lane XOR-butterfly max without LDS: v_permlane16_b32 (4 VALU ops).
// ---------------------------------------------------------------------------
#if __has_builtin(__builtin_amdgcn_permlane16)
__device__ inline float pl16(float x, unsigned s0, unsigned s1) {
  unsigned i = (unsigned)__float_as_int(x);
  unsigned r = __builtin_amdgcn_permlane16(i, i, s0, s1, false, false);
  return __int_as_float((int)r);
}
__device__ inline float rowmax16(float x) {
  x = fmaxf(x, pl16(x, 0x67452301u, 0xEFCDAB89u));  // xor 1
  x = fmaxf(x, pl16(x, 0x45670123u, 0xCDEF89ABu));  // xor 2
  x = fmaxf(x, pl16(x, 0x32107654u, 0xBA98FEDCu));  // xor 4
  x = fmaxf(x, pl16(x, 0xFEDCBA98u, 0x76543210u));  // xor 8
  return x;
}
#else
__device__ inline float rowmax16(float x) {
  for (int m = 1; m < 16; m <<= 1) x = fmaxf(x, __shfl_xor(x, m, 32));
  return x;
}
#endif

// mish(x) = x * tanh(softplus(x)) = x * ((1+e^x)^2 - 1) / ((1+e^x)^2 + 1)
__device__ inline float mish_f(float x) {
  float ex = __expf(fminf(x, 20.f));
  float t  = 1.f + ex;
  float t2 = t * t;
  return x * (t2 - 1.f) * __builtin_amdgcn_rcpf(t2 + 1.f);
}

// ---------------------------------------------------------------------------
// Elementwise f32 -> bf16 convert, 4-wide
// ---------------------------------------------------------------------------
__global__ void cvt_bf16_kernel(const float* __restrict__ in,
                                __bf16* __restrict__ out, long n4) {
  long i = (long)blockIdx.x * blockDim.x + threadIdx.x;
  long s = (long)gridDim.x * blockDim.x;
  for (; i < n4; i += s) {
    float4 v = ((const float4*)in)[i];
    v4bf o;
    o[0] = (__bf16)v.x; o[1] = (__bf16)v.y;
    o[2] = (__bf16)v.z; o[3] = (__bf16)v.w;
    ((v4bf*)out)[i] = o;
  }
}

// ---------------------------------------------------------------------------
// GEMM: C[M,N] = A[M,K](bf16) @ W[N,K]^T(bf16) + bias, optional mish.
// 128x128 C tile, 8 waves (4x2), 32x64 per wave. K in 64-chunks, ping-pong
// double-buffered LDS, next stage's global loads issued during WMMAs.
// ---------------------------------------------------------------------------
#define GTM 128
#define GTN 128
#define GTK 64

template <int ACT, bool OUTF, bool OUTB>
__global__ __launch_bounds__(256) void gemm_bf16_kernel(
    const __bf16* __restrict__ A, const __bf16* __restrict__ W,
    const float* __restrict__ bias, float* __restrict__ Cf,
    __bf16* __restrict__ Cb, int M, int N, int K) {
  __shared__ __bf16 As[2][GTM * GTK];
  __shared__ __bf16 Ws[2][GTN * GTK];

  const int m0 = blockIdx.x * GTM;
  const int n0 = blockIdx.y * GTN;
  const int tid = threadIdx.x;
  const int wave = tid >> 5;
  const int wr = wave >> 1;
  const int wc = wave & 1;

  v8f acc[2][4];
#pragma unroll
  for (int i = 0; i < 2; ++i)
#pragma unroll
    for (int j = 0; j < 4; ++j)
#pragma unroll
      for (int e = 0; e < 8; ++e) acc[i][j][e] = 0.f;

  const int lr = tid >> 1;
  const int lc = (tid & 1) * 32;

  const __bf16* ga = A + (size_t)(m0 + lr) * K + lc;
  const __bf16* gw = W + (size_t)(n0 + lr) * K + lc;
  const int S = K / GTK;

  uint4 ra[4], rw[4];
#pragma unroll
  for (int u = 0; u < 4; ++u) {
    ra[u] = *(const uint4*)(ga + u * 8);
    rw[u] = *(const uint4*)(gw + u * 8);
  }
#pragma unroll
  for (int u = 0; u < 4; ++u) {
    *(uint4*)&As[0][lr * GTK + lc + u * 8] = ra[u];
    *(uint4*)&Ws[0][lr * GTK + lc + u * 8] = rw[u];
  }

  for (int s = 0; s < S; ++s) {
    __syncthreads();
    if (s + 1 < S) {
      const __bf16* ga2 = ga + (size_t)(s + 1) * GTK;
      const __bf16* gw2 = gw + (size_t)(s + 1) * GTK;
#pragma unroll
      for (int u = 0; u < 4; ++u) {
        ra[u] = *(const uint4*)(ga2 + u * 8);
        rw[u] = *(const uint4*)(gw2 + u * 8);
      }
    }
    const __bf16* as  = &As[s & 1][0];
    const __bf16* wsm = &Ws[s & 1][0];
#pragma unroll
    for (int kc = 0; kc < 2; ++kc) {
      v16bf af[2], bf2[4];
#pragma unroll
      for (int i = 0; i < 2; ++i)
        af[i] = frag_ld(as + (wr * 32 + i * 16) * GTK + kc * 32, GTK);
#pragma unroll
      for (int j = 0; j < 4; ++j)
        bf2[j] = frag_ld(wsm + (wc * 64 + j * 16) * GTK + kc * 32, GTK);
#pragma unroll
      for (int i = 0; i < 2; ++i)
#pragma unroll
        for (int j = 0; j < 4; ++j)
          acc[i][j] = wmma_bf16(af[i], bf2[j], acc[i][j]);
    }
    if (s + 1 < S) {
      __bf16* an = &As[(s + 1) & 1][0];
      __bf16* wn = &Ws[(s + 1) & 1][0];
#pragma unroll
      for (int u = 0; u < 4; ++u) {
        *(uint4*)&an[lr * GTK + lc + u * 8] = ra[u];
        *(uint4*)&wn[lr * GTK + lc + u * 8] = rw[u];
      }
    }
  }

  const int lane = tid & 31;
  const int col  = lane & 15;
  const int rofs = (lane >> 4) * 8;
#pragma unroll
  for (int j = 0; j < 4; ++j) {
    const int gn = n0 + wc * 64 + j * 16 + col;
    const float bia = bias[gn];
#pragma unroll
    for (int i = 0; i < 2; ++i) {
#pragma unroll
      for (int e = 0; e < 8; ++e) {
        const int gm = m0 + wr * 32 + i * 16 + rofs + e;
        float v = acc[i][j][e] + bia;
        if (ACT == 1) v = mish_f(v);
        const size_t idx = (size_t)gm * N + gn;
        if (OUTF) Cf[idx] = v;
        if (OUTB) Cb[idx] = (__bf16)v;
      }
    }
  }
}

// ---------------------------------------------------------------------------
// Per-head RMSNorm (hd = 64): one wave per (token, head), f32 in -> bf16 out.
// ---------------------------------------------------------------------------
__global__ __launch_bounds__(256) void rms_head_kernel(
    const float* __restrict__ in, const float* __restrict__ w,
    __bf16* __restrict__ out, int total) {
  const int vec = blockIdx.x * 8 + (threadIdx.x >> 5);
  if (vec >= total) return;
  const int lane = threadIdx.x & 31;
  const float* p = in + (size_t)vec * 64;
  const float a = p[lane], b = p[lane + 32];
  float ss = a * a + b * b;
#pragma unroll
  for (int m = 1; m < 32; m <<= 1) ss += __shfl_xor(ss, m, 32);
  const float inv = rsqrtf(ss * (1.f / 64.f) + 1e-6f);
  __bf16* o = out + (size_t)vec * 64;
  o[lane]      = (__bf16)(a * inv * w[lane]);
  o[lane + 32] = (__bf16)(b * inv * w[lane + 32]);
}

// ---------------------------------------------------------------------------
// Flash attention with additive bias; all dims compile-time so the bias row
// offsets (j*4096 B) fold into immediate offsets of a single base address.
// One block = 128 q-rows of one (b,h): 8 waves x 16 rows. Ping-pong
// double-buffered 32-key LDS tiles (V transposed at store). S = Q K^T and
// O += P V via WMMA; P row sums via ones-matrix WMMA; row max via permlane16.
// ---------------------------------------------------------------------------
__global__ __launch_bounds__(256) void attn_kernel(
    const __bf16* __restrict__ qb, const __bf16* __restrict__ kb,
    const __bf16* __restrict__ vb, const float* __restrict__ bias,
    __bf16* __restrict__ outp) {
  constexpr int T = 1024, D = 1024, H = 16;
  __shared__ __bf16 Ks[2][32 * 64];
  __shared__ __bf16 Vt[2][64 * 32];
  __shared__ __bf16 Ps[8][16 * 32];

  const int bh = blockIdx.y;
  const int b = bh / H, h = bh % H;
  const int wave = threadIdx.x >> 5;
  const int lane = threadIdx.x & 31;
  const int q0 = blockIdx.x * 128 + wave * 16;

  const __bf16* qptr = qb + ((size_t)b * T + q0) * D + h * 64;
  v16bf qf[2];
  qf[0] = frag_ld(qptr, D);
  qf[1] = frag_ld(qptr + 32, D);

  v16bf ones;
#pragma unroll
  for (int e = 0; e < 16; ++e) ones[e] = (__bf16)1.f;

  float rmax[8], rsum[8];
  v8f o[4];
#pragma unroll
  for (int j = 0; j < 8; ++j) { rmax[j] = -1e30f; rsum[j] = 0.f; }
#pragma unroll
  for (int nt = 0; nt < 4; ++nt)
#pragma unroll
    for (int e = 0; e < 8; ++e) o[nt][e] = 0.f;

  const float scale = 0.125f;  // 1/sqrt(64)
  const int half8 = (lane >> 4) * 8;
  const int col = lane & 15;

  // single bias base; per-row offsets j*T*4 B and +64 B become immediates
  const float* bpbase = bias + ((size_t)bh * T + q0 + half8) * T + col;

  const int r = threadIdx.x >> 3;
  const int c = (threadIdx.x & 7) * 8;
  const __bf16* kg = kb + ((size_t)b * T + r) * D + h * 64 + c;
  const __bf16* vg = vb + ((size_t)b * T + r) * D + h * 64 + c;

  union { uint4 u; __bf16 e[8]; } krg, vrg;
  krg.u = *(const uint4*)kg;
  vrg.u = *(const uint4*)vg;
  *(uint4*)&Ks[0][r * 64 + c] = krg.u;
#pragma unroll
  for (int i = 0; i < 8; ++i) Vt[0][(c + i) * 32 + r] = vrg.e[i];

  constexpr int NIT = T / 32;
  for (int it = 0; it < NIT; ++it) {
    __syncthreads();
    if (it + 1 < NIT) {  // issue next tile's global loads early
      krg.u = *(const uint4*)(kg + (size_t)(it + 1) * 32 * D);
      vrg.u = *(const uint4*)(vg + (size_t)(it + 1) * 32 * D);
    }
    const __bf16* ks = &Ks[it & 1][0];
    const __bf16* vt = &Vt[it & 1][0];

    v8f s[2];
#pragma unroll
    for (int t = 0; t < 2; ++t) {
      v8f z;
#pragma unroll
      for (int e = 0; e < 8; ++e) z[e] = 0.f;
      z = wmma_bf16(qf[0], frag_ld(ks + (t * 16) * 64, 64), z);
      z = wmma_bf16(qf[1], frag_ld(ks + (t * 16) * 64 + 32, 64), z);
      s[t] = z;
    }

    const float* bp = bpbase + it * 32;
    float corr[8];
#pragma unroll
    for (int j = 0; j < 8; ++j) {
      const float s0 = s[0][j] * scale + NT_LOAD(bp + j * T);
      const float s1 = s[1][j] * scale + NT_LOAD(bp + j * T + 16);
      const float mx = rowmax16(fmaxf(s0, s1));
      const float nm = fmaxf(rmax[j], mx);
      const float c0 = __expf(rmax[j] - nm);
      rmax[j] = nm;
      corr[j] = c0;
      const float p0 = __expf(s0 - nm);
      const float p1 = __expf(s1 - nm);
      Ps[wave][(j + half8) * 32 + col]      = (__bf16)p0;
      Ps[wave][(j + half8) * 32 + 16 + col] = (__bf16)p1;
    }
#pragma unroll
    for (int nt = 0; nt < 4; ++nt)
#pragma unroll
      for (int j = 0; j < 8; ++j) o[nt][j] *= corr[j];

    const v16bf pa = frag_ld(&Ps[wave][0], 32);
    v8f sums;
#pragma unroll
    for (int e = 0; e < 8; ++e) sums[e] = 0.f;
    sums = wmma_bf16(pa, ones, sums);
#pragma unroll
    for (int j = 0; j < 8; ++j) rsum[j] = rsum[j] * corr[j] + sums[j];

#pragma unroll
    for (int nt = 0; nt < 4; ++nt)
      o[nt] = wmma_bf16(pa, frag_ld(vt + (nt * 16) * 32, 32), o[nt]);

    if (it + 1 < NIT) {  // fill the other buffer
      *(uint4*)&Ks[(it + 1) & 1][r * 64 + c] = krg.u;
#pragma unroll
      for (int i = 0; i < 8; ++i) Vt[(it + 1) & 1][(c + i) * 32 + r] = vrg.e[i];
    }
  }

#pragma unroll
  for (int j = 0; j < 8; ++j) {
    const int row = q0 + j + half8;
    const float invs = __builtin_amdgcn_rcpf(rsum[j]);
#pragma unroll
    for (int nt = 0; nt < 4; ++nt) {
      const int d = h * 64 + nt * 16 + col;
      outp[((size_t)b * T + row) * D + d] = (__bf16)(o[nt][j] * invs);
    }
  }
}

// ---------------------------------------------------------------------------
// Fused residual + LayerNorm over D=1024. One block per token.
// ---------------------------------------------------------------------------
template <bool OUTBF>
__global__ __launch_bounds__(256) void ln_fuse_kernel(
    const float* __restrict__ a, const float* __restrict__ r,
    const float* __restrict__ w, const float* __restrict__ bb,
    float* __restrict__ outf, __bf16* __restrict__ outb, int D) {
  const int token = blockIdx.x;
  const int tid = threadIdx.x;
  const float* pa = a + (size_t)token * D;
  const float* pr = r + (size_t)token * D;
  float x[4], s = 0.f, s2 = 0.f;
#pragma unroll
  for (int i = 0; i < 4; ++i) {
    x[i] = pa[tid + i * 256] + pr[tid + i * 256];
    s += x[i];
    s2 += x[i] * x[i];
  }
  __shared__ float red[2][8];
#pragma unroll
  for (int m = 1; m < 32; m <<= 1) {
    s  += __shfl_xor(s, m, 32);
    s2 += __shfl_xor(s2, m, 32);
  }
  if ((tid & 31) == 0) { red[0][tid >> 5] = s; red[1][tid >> 5] = s2; }
  __syncthreads();
  s = 0.f; s2 = 0.f;
#pragma unroll
  for (int i = 0; i < 8; ++i) { s += red[0][i]; s2 += red[1][i]; }
  const float mu = s / D;
  const float var = s2 / D - mu * mu;
  const float inv = rsqrtf(var + 1e-5f);
#pragma unroll
  for (int i = 0; i < 4; ++i) {
    const int d = tid + i * 256;
    const float y = (x[i] - mu) * inv * w[d] + bb[d];
    const size_t idx = (size_t)token * D + d;
    outf[idx] = y;
    if (OUTBF) outb[idx] = (__bf16)y;
  }
}

// ---------------------------------------------------------------------------
// Host-side orchestration
// ---------------------------------------------------------------------------
extern "C" void kernel_launch(void* const* d_in, const int* in_sizes, int n_in,
                              void* d_out, int out_size, void* d_ws,
                              size_t ws_size, hipStream_t stream) {
  const int B = 8, T = 1024, D = 1024, H = 16, FF = 2048;
  const int M = B * T;

  const float* x   = (const float*)d_in[0];
  const float* sb  = (const float*)d_in[1];
  const float* Wq  = (const float*)d_in[2];
  const float* bq  = (const float*)d_in[3];
  const float* Wk  = (const float*)d_in[4];
  const float* bk  = (const float*)d_in[5];
  const float* Wv  = (const float*)d_in[6];
  const float* bv  = (const float*)d_in[7];
  const float* Wo  = (const float*)d_in[8];
  const float* bo  = (const float*)d_in[9];
  const float* qn  = (const float*)d_in[10];
  const float* kn  = (const float*)d_in[11];
  const float* l1w = (const float*)d_in[12];
  const float* l1b = (const float*)d_in[13];
  const float* W1  = (const float*)d_in[14];
  const float* b1  = (const float*)d_in[15];
  const float* W2  = (const float*)d_in[16];
  const float* b2  = (const float*)d_in[17];
  const float* l2w = (const float*)d_in[18];
  const float* l2b = (const float*)d_in[19];

  char* ws = (char*)d_ws;
  const size_t MB = 1024ull * 1024ull;
  __bf16* xb    = (__bf16*)(ws + 0 * MB);
  __bf16* wqb   = (__bf16*)(ws + 16 * MB);
  __bf16* wkb   = (__bf16*)(ws + 18 * MB);
  __bf16* wvb   = (__bf16*)(ws + 20 * MB);
  __bf16* wob   = (__bf16*)(ws + 22 * MB);
  __bf16* w1b   = (__bf16*)(ws + 24 * MB);
  __bf16* w2b   = (__bf16*)(ws + 28 * MB);
  float*  qf    = (float*)(ws + 32 * MB);   // later reused as proj_f
  float*  kf    = (float*)(ws + 64 * MB);   // later reused as ln1_f
  float*  ffn2f = (float*)(ws + 96 * MB);
  __bf16* qbb   = (__bf16*)(ws + 128 * MB);
  __bf16* kbb   = (__bf16*)(ws + 144 * MB);
  __bf16* vbb   = (__bf16*)(ws + 160 * MB);
  __bf16* attb  = (__bf16*)(ws + 176 * MB);
  __bf16* ln1bf = (__bf16*)(ws + 192 * MB);
  __bf16* midb  = (__bf16*)(ws + 208 * MB);
  float*  projf = qf;
  float*  ln1f  = kf;

  cvt_bf16_kernel<<<2048, 256, 0, stream>>>(x, xb, (long)M * D / 4);
  cvt_bf16_kernel<<<512, 256, 0, stream>>>(Wq, wqb, (long)D * D / 4);
  cvt_bf16_kernel<<<512, 256, 0, stream>>>(Wk, wkb, (long)D * D / 4);
  cvt_bf16_kernel<<<512, 256, 0, stream>>>(Wv, wvb, (long)D * D / 4);
  cvt_bf16_kernel<<<512, 256, 0, stream>>>(Wo, wob, (long)D * D / 4);
  cvt_bf16_kernel<<<1024, 256, 0, stream>>>(W1, w1b, (long)FF * D / 4);
  cvt_bf16_kernel<<<1024, 256, 0, stream>>>(W2, w2b, (long)D * FF / 4);

  dim3 g1(M / GTM, D / GTN);
  gemm_bf16_kernel<0, true, false><<<g1, 256, 0, stream>>>(xb, wqb, bq, qf, nullptr, M, D, D);
  gemm_bf16_kernel<0, true, false><<<g1, 256, 0, stream>>>(xb, wkb, bk, kf, nullptr, M, D, D);
  gemm_bf16_kernel<0, false, true><<<g1, 256, 0, stream>>>(xb, wvb, bv, nullptr, vbb, M, D, D);

  rms_head_kernel<<<(M * H) / 8, 256, 0, stream>>>(qf, qn, qbb, M * H);
  rms_head_kernel<<<(M * H) / 8, 256, 0, stream>>>(kf, kn, kbb, M * H);

  attn_kernel<<<dim3(T / 128, B * H), 256, 0, stream>>>(qbb, kbb, vbb, sb, attb);

  gemm_bf16_kernel<0, true, false><<<g1, 256, 0, stream>>>(attb, wob, bo, projf, nullptr, M, D, D);
  ln_fuse_kernel<true><<<M, 256, 0, stream>>>(x, projf, l1w, l1b, ln1f, ln1bf, D);

  dim3 g2(M / GTM, FF / GTN);
  gemm_bf16_kernel<1, false, true><<<g2, 256, 0, stream>>>(ln1bf, w1b, b1, nullptr, midb, M, FF, D);
  gemm_bf16_kernel<0, true, false><<<g1, 256, 0, stream>>>(midb, w2b, b2, ffn2f, nullptr, M, D, FF);

  ln_fuse_kernel<false><<<M, 256, 0, stream>>>(ln1f, ffn2f, l2w, l2b,
                                               (float*)d_out, nullptr, D);
}
